// AnalyticGaussianVelocity_40913858462198
// MI455X (gfx1250) — matrix operand order repaired and checked by reading, not compiled, not run on hardware
//
#include <hip/hip_runtime.h>
#include <math.h>

typedef __attribute__((ext_vector_type(16))) _Float16 v16h;
typedef __attribute__((ext_vector_type(4)))  _Float16 v4h;
typedef __attribute__((ext_vector_type(8)))  float    v8f;

#define D_DIM 512
#define NT    64            // dataset rows per flash tile

// swizzle within a 32-element K block into HW A-fragment order:
// [0..7, 16..23, 8..15, 24..31]
__device__ __forceinline__ int kswz(int kk) {        // kk in [0,32)
  int s = kk >> 3;
  int spos = (s == 1) ? 2 : (s == 2) ? 1 : s;
  return spos * 8 + (kk & 7);
}

// ---------------------------------------------------------------------------
// Kernel 1: f32 -> f16 row-major copy (optional) + row norms (optional).
// ---------------------------------------------------------------------------
__global__ __launch_bounds__(256) void cvt_norm_kernel(const float* __restrict__ src,
                                                       _Float16* __restrict__ dst16,
                                                       float* __restrict__ norms,
                                                       int rows) {
  int row  = blockIdx.x * 8 + (threadIdx.x >> 5);
  int lane = threadIdx.x & 31;
  if (row >= rows) return;
  const float4* p = reinterpret_cast<const float4*>(src + (size_t)row * D_DIM);
  v4h* q = dst16 ? reinterpret_cast<v4h*>(dst16 + (size_t)row * D_DIM) : nullptr;
  float s = 0.0f;
#pragma unroll
  for (int i = 0; i < 4; ++i) {
    float4 v = p[lane + 32 * i];
    if (q) {
      v4h h = {(_Float16)v.x, (_Float16)v.y, (_Float16)v.z, (_Float16)v.w};
      q[lane + 32 * i] = h;
    }
    s += v.x * v.x + v.y * v.y + v.z * v.z + v.w * v.w;
  }
  if (norms) {
#pragma unroll
    for (int off = 16; off > 0; off >>= 1) s += __shfl_xor(s, off, 32);
    if (lane == 0) norms[row] = s;
  }
}

// ---------------------------------------------------------------------------
// Kernel 1b: f32 -> f16 row-major copy, contiguous K order (for x_t).
// ---------------------------------------------------------------------------
__global__ __launch_bounds__(256) void xswz_kernel(const float* __restrict__ src,
                                                   _Float16* __restrict__ dst,
                                                   int rows) {
  int row  = blockIdx.x * 8 + (threadIdx.x >> 5);
  int lane = threadIdx.x & 31;
  if (row >= rows) return;
  const float4* p = reinterpret_cast<const float4*>(src + (size_t)row * D_DIM);
  _Float16* drow = dst + (size_t)row * D_DIM;
#pragma unroll
  for (int i = 0; i < 4; ++i) {
    int k = (lane + 32 * i) * 4;
    float4 v = p[lane + 32 * i];
    int pos = k;                          // contiguous K order
    v4h h = {(_Float16)v.x, (_Float16)v.y, (_Float16)v.z, (_Float16)v.w};
    *reinterpret_cast<v4h*>(drow + pos) = h;
  }
}

// ---------------------------------------------------------------------------
// Kernel 1c: 32x32-tiled transpose, f32 (N x 512) -> f16 (512 x N_pad),
// zero-filled for n >= N.
// ---------------------------------------------------------------------------
__global__ __launch_bounds__(256) void transpose_kernel(const float* __restrict__ src,
                                                        _Float16* __restrict__ dst,
                                                        int N, int N_pad) {
  __shared__ float tile[32][33];
  const int nb = blockIdx.x * 32;
  const int db = blockIdx.y * 32;
  const int tx = threadIdx.x & 31;
  const int ty = threadIdx.x >> 5;
#pragma unroll
  for (int r = 0; r < 32; r += 8) {
    int n = nb + ty + r;
    tile[ty + r][tx] = (n < N) ? src[(size_t)n * D_DIM + db + tx] : 0.0f;
  }
  __syncthreads();
#pragma unroll
  for (int r = 0; r < 32; r += 8) {
    int d = db + ty + r;
    dst[(size_t)d * N_pad + nb + tx] = (_Float16)tile[tx][ty + r];
  }
}

// ---------------------------------------------------------------------------
// Kernel 2: fused flash-style softmax-weighted velocity.
//   logit_bn = alpha_b*<x_b,d_n> + beta_b*||d_n||^2  (row const dropped)
//   v_b      = (-1/b_t) x_b + (1 + a/b) * sum_n softmax(logit)_bn d_n
// Workgroup = 16 rows of B, 8 wave32, N-tile = 64.
// F16=true : all fragments are contiguous 32B v16h loads (row-major x16,
//            row-major ds16 for phase A, transposed dsT for phase C).
// ---------------------------------------------------------------------------
template <bool F16>
__global__ __launch_bounds__(256) void flash_vel_kernel(
    const float* __restrict__ x32, const float* __restrict__ t_arr,
    const float* __restrict__ ds32, const void* __restrict__ x16v,
    const void* __restrict__ ds16v, const void* __restrict__ dsTv,
    const float* __restrict__ dnorm, float* __restrict__ out,
    int N, int N_pad) {

  const _Float16* x16sw = (const _Float16*)x16v;   // row-major rows
  const _Float16* ds16  = (const _Float16*)ds16v;  // row-major
  const _Float16* dsT   = (const _Float16*)dsTv;   // transposed, padded

  __shared__ float S_lds[16][NT + 1];                        // logits
  __shared__ __align__(32) _Float16 P16_lds[16][80];         // probs, n order
  __shared__ float alpha_s[16], beta_s[16], selfc_s[16], datac_s[16];
  __shared__ float rowmax_s[16], rowsum_s[16], cf_s[16], inv_s[16];

  const int tid   = threadIdx.x;
  const int wid   = tid >> 5;
  const int lane  = tid & 31;
  const int lhalf = lane >> 4;
  const int l15   = lane & 15;
  const int b0    = blockIdx.x * 16;
  const int dbase = wid * 64;

  if (tid < 16) {
    float t = t_arr[b0 + tid];
    float a = t, b = 1.0f - t;
    alpha_s[tid]  = a / (b * b);
    beta_s[tid]   = -(a * a) / (2.0f * b * b);
    selfc_s[tid]  = -1.0f / b;
    datac_s[tid]  = 1.0f + a / b;
    rowmax_s[tid] = -INFINITY;
    rowsum_s[tid] = 0.0f;
  }
  __syncthreads();

  v8f acc[4] = {v8f{}, v8f{}, v8f{}, v8f{}};

  for (int n0 = 0; n0 < N; n0 += NT) {
    // ---- phase A: waves 0..3 each compute a 16x16 logit tile --------------
    if (wid < 4) {
      const int ncol  = wid * 16 + l15;
      const int nglob = n0 + ncol;
      const int nr    = nglob < N ? nglob : N - 1;
      v8f dotc = {};
      if constexpr (F16) {
        const _Float16* arow = x16sw + (size_t)(b0 + l15) * D_DIM;
        const _Float16* brow = ds16 + (size_t)nr * D_DIM;
        {  // prefetch next tile's row for this lane
          int np = nr + NT; if (np >= N) np = N - 1;
          __builtin_prefetch(ds16 + (size_t)np * D_DIM, 0, 1);
        }
#pragma unroll 4
        for (int kb = 0; kb < D_DIM; kb += 32) {
          // A row-major: contiguous 32B per lane-half; B row-major: same K order as A
          // (16 consecutive K per lane-half) is naturally contiguous.
          v16h afrag = *reinterpret_cast<const v16h*>(arow + kb + lhalf * 16);
          v16h bfrag = *reinterpret_cast<const v16h*>(brow + kb + lhalf * 16);
          dotc = __builtin_amdgcn_wmma_f32_16x16x32_f16(
              false, afrag, false, bfrag, (short)0, dotc, false, false);
        }
      } else {
        const float* arow = x32 + (size_t)(b0 + l15) * D_DIM;
        const float* brow = ds32 + (size_t)nr * D_DIM;
#pragma unroll 2
        for (int kb = 0; kb < D_DIM; kb += 32) {
          v16h afrag, bfrag;
          const float4* pa0 = reinterpret_cast<const float4*>(arow + kb + lhalf * 16);
          const float4* pa1 = reinterpret_cast<const float4*>(arow + kb + lhalf * 16 + 8);
          const float4* pb  = reinterpret_cast<const float4*>(brow + kb + lhalf * 16);
          float4 qa0 = pa0[0], qa1 = pa0[1], qa2 = pa1[0], qa3 = pa1[1];
          float4 qb0 = pb[0], qb1 = pb[1], qb2 = pb[2], qb3 = pb[3];
          afrag[0]=(_Float16)qa0.x; afrag[1]=(_Float16)qa0.y; afrag[2]=(_Float16)qa0.z; afrag[3]=(_Float16)qa0.w;
          afrag[4]=(_Float16)qa1.x; afrag[5]=(_Float16)qa1.y; afrag[6]=(_Float16)qa1.z; afrag[7]=(_Float16)qa1.w;
          afrag[8]=(_Float16)qa2.x; afrag[9]=(_Float16)qa2.y; afrag[10]=(_Float16)qa2.z; afrag[11]=(_Float16)qa2.w;
          afrag[12]=(_Float16)qa3.x; afrag[13]=(_Float16)qa3.y; afrag[14]=(_Float16)qa3.z; afrag[15]=(_Float16)qa3.w;
          bfrag[0]=(_Float16)qb0.x; bfrag[1]=(_Float16)qb0.y; bfrag[2]=(_Float16)qb0.z; bfrag[3]=(_Float16)qb0.w;
          bfrag[4]=(_Float16)qb1.x; bfrag[5]=(_Float16)qb1.y; bfrag[6]=(_Float16)qb1.z; bfrag[7]=(_Float16)qb1.w;
          bfrag[8]=(_Float16)qb2.x; bfrag[9]=(_Float16)qb2.y; bfrag[10]=(_Float16)qb2.z; bfrag[11]=(_Float16)qb2.w;
          bfrag[12]=(_Float16)qb3.x; bfrag[13]=(_Float16)qb3.y; bfrag[14]=(_Float16)qb3.z; bfrag[15]=(_Float16)qb3.w;
          dotc = __builtin_amdgcn_wmma_f32_16x16x32_f16(
              false, afrag, false, bfrag, (short)0, dotc, false, false);
        }
      }
      const float dn    = dnorm[nr];
      const bool  valid = (nglob < N);
#pragma unroll
      for (int j = 0; j < 8; ++j) {
        int m = j + (lhalf << 3);            // C layout: VGPR j -> rows j / j+8
        S_lds[m][ncol] = valid ? (alpha_s[m] * dotc[j] + beta_s[m] * dn) : -INFINITY;
      }
    }
    __syncthreads();

    // ---- phase B: online softmax (lanes 0..15); write P in n order ----------
    if (tid < 16) {
      const int m = tid;
      float tmax = -INFINITY;
#pragma unroll
      for (int n = 0; n < NT; ++n) tmax = fmaxf(tmax, S_lds[m][n]);
      float mold = rowmax_s[m];
      float mnew = fmaxf(mold, tmax);
      float cf   = __expf(mold - mnew);
      float sum  = 0.0f;
#pragma unroll
      for (int n = 0; n < NT; ++n) {
        float p = __expf(S_lds[m][n] - mnew);
        sum += p;
        P16_lds[m][n] = (_Float16)p;
      }
      rowsum_s[m] = rowsum_s[m] * cf + sum;
      rowmax_s[m] = mnew;
      cf_s[m]     = cf;
    }
    __syncthreads();

    // ---- phase C: rescale acc, acc += P(16x64) x Dtile(64 x 64-slice) -----
    v16h pfrag[2];
#pragma unroll
    for (int h = 0; h < 2; ++h)
      pfrag[h] = *reinterpret_cast<const v16h*>(&P16_lds[l15][h * 32 + lhalf * 16]);

#pragma unroll
    for (int j = 0; j < 8; ++j) {
      float cf = cf_s[j + (lhalf << 3)];
      acc[0][j] *= cf; acc[1][j] *= cf; acc[2][j] *= cf; acc[3][j] *= cf;
    }
#pragma unroll
    for (int c = 0; c < 4; ++c) {
      const int dcol = dbase + c * 16 + l15;
#pragma unroll
      for (int h = 0; h < 2; ++h) {
        v16h bfrag;
        if constexpr (F16) {
          // transposed copy: 16 consecutive n at fixed d, zero-padded -> one load
          bfrag = *reinterpret_cast<const v16h*>(
              dsT + (size_t)dcol * N_pad + n0 + h * 32 + lhalf * 16);
        } else {
#pragma unroll
          for (int e = 0; e < 16; ++e) {
            int nl = h * 32 + lhalf * 16 + e;
            int nr = n0 + nl;
            if (nr >= N) nr = N - 1;          // p==0 there, value irrelevant
            bfrag[e] = (_Float16)ds32[(size_t)nr * D_DIM + dcol];
          }
        }
        acc[c] = __builtin_amdgcn_wmma_f32_16x16x32_f16(
            false, pfrag[h], false, bfrag, (short)0, acc[c], false, false);
      }
    }
    __syncthreads();
  }

  if (tid < 16) inv_s[tid] = 1.0f / rowsum_s[tid];
  __syncthreads();

  // ---- epilogue: v = selfc*x + datac*acc/l --------------------------------
#pragma unroll
  for (int c = 0; c < 4; ++c) {
#pragma unroll
    for (int j = 0; j < 8; ++j) {
      const int m   = j + (lhalf << 3);
      const int d   = dbase + c * 16 + l15;
      const int row = b0 + m;
      const float xv = x32[(size_t)row * D_DIM + d];
      out[(size_t)row * D_DIM + d] =
          selfc_s[m] * xv + datac_s[m] * acc[c][j] * inv_s[m];
    }
  }
}

// ---------------------------------------------------------------------------
extern "C" void kernel_launch(void* const* d_in, const int* in_sizes, int n_in,
                              void* d_out, int out_size, void* d_ws, size_t ws_size,
                              hipStream_t stream) {
  const float* x_t     = (const float*)d_in[0];
  const float* t_arr   = (const float*)d_in[1];
  const float* dataset = (const float*)d_in[2];
  const int B = in_sizes[1];            // 1024
  const int N = in_sizes[2] / D_DIM;    // 50000
  const int N_pad = (N + 63) & ~63;
  float* out = (float*)d_out;

  const size_t norm_b = ((size_t)N * 4 + 255) & ~(size_t)255;
  const size_t ds16_b = ((size_t)N * D_DIM * 2 + 255) & ~(size_t)255;
  const size_t dsT_b  = ((size_t)N_pad * D_DIM * 2 + 255) & ~(size_t)255;
  const size_t x16_b  = ((size_t)B * D_DIM * 2 + 255) & ~(size_t)255;

  float* dnorm = (float*)d_ws;

  if (ws_size >= norm_b + ds16_b + dsT_b + x16_b) {
    // fast path: f16 row-major + f16 transposed + row-major f16 x_t
    _Float16* ds16 = (_Float16*)((char*)d_ws + norm_b);
    _Float16* dsT  = (_Float16*)((char*)d_ws + norm_b + ds16_b);
    _Float16* x16  = (_Float16*)((char*)d_ws + norm_b + ds16_b + dsT_b);
    cvt_norm_kernel<<<(N + 7) / 8, 256, 0, stream>>>(dataset, ds16, dnorm, N);
    xswz_kernel<<<(B + 7) / 8, 256, 0, stream>>>(x_t, x16, B);
    dim3 tg(N_pad / 32, D_DIM / 32);
    transpose_kernel<<<tg, 256, 0, stream>>>(dataset, dsT, N, N_pad);
    flash_vel_kernel<true><<<B / 16, 256, 0, stream>>>(
        x_t, t_arr, dataset, x16, ds16, dsT, dnorm, out, N, N_pad);
  } else {
    // fallback: norms only in workspace, convert f32->f16 on the fly
    cvt_norm_kernel<<<(N + 7) / 8, 256, 0, stream>>>(dataset, nullptr, dnorm, N);
    flash_vel_kernel<false><<<B / 16, 256, 0, stream>>>(
        x_t, t_arr, dataset, nullptr, nullptr, nullptr, dnorm, out, N, N_pad);
  }
}
